// HeteroGNNLayer_1709396984163
// MI455X (gfx1250) — matrix operand order repaired and checked by reading, not compiled, not run on hardware
//
#include <hip/hip_runtime.h>

#define HID 256
#define NH  8
#define SLOPE 0.2f

typedef __attribute__((ext_vector_type(16))) __bf16 v16bf;
typedef __attribute__((ext_vector_type(8)))  __bf16 v8bf;
typedef __attribute__((ext_vector_type(8)))  float  v8f;

__device__ __forceinline__ int f2ord(float f){ int i=__float_as_int(f); return (i>=0)? i : (i ^ 0x7fffffff); }
__device__ __forceinline__ float ord2f(int i){ return __int_as_float((i>=0)? i : (i ^ 0x7fffffff)); }

// ---------------- helpers ----------------

__global__ void k_fill_u32(unsigned* __restrict__ p, unsigned v, int n){
  int i = blockIdx.x*blockDim.x + threadIdx.x;
  if (i < n) p[i] = v;
}

__global__ void k_f2bf(const float* __restrict__ x, __bf16* __restrict__ y, int n){
  int i = blockIdx.x*blockDim.x + threadIdx.x;
  if (i < n) y[i] = (__bf16)x[i];
}

// W (K x N) fp32 -> Wt (N x K) bf16, 256x256
__global__ void k_wtr(const float* __restrict__ w, __bf16* __restrict__ wt){
  int i = blockIdx.x*blockDim.x + threadIdx.x;
  int k = i >> 8, n = i & 255;
  wt[n*HID + k] = (__bf16)w[k*HID + n];
}

// acc = x + b1 (+ b2)   (residual + conv biases seeded before scatter-add)
__global__ void k_init_acc(const float* __restrict__ x, const float* __restrict__ b1,
                           const float* __restrict__ b2, float* __restrict__ acc, int total){
  int i = blockIdx.x*blockDim.x + threadIdx.x;
  if (i >= total) return;
  int c = i & (HID-1);
  float v = x[i] + b1[c];
  if (b2) v += b2[c];
  acc[i] = v;
}

// ---------------- WMMA GEMM:  H = X(bf16) @ Wt^T, fused attention-logit epilogue ----------------
// Each wave computes one 16x16 f32 tile via 8x v_wmma_f32_16x16x32_bf16.
// Optional: store H; optional: al_s[m,h] += sum_c D[m,n]*a_s[n]; same for al_d.
__global__ void __launch_bounds__(256)
k_gemm(const __bf16* __restrict__ X, const __bf16* __restrict__ Wt,
       float* __restrict__ Hout,
       const float* __restrict__ avec_s, float* __restrict__ al_s,
       const float* __restrict__ avec_d, float* __restrict__ al_d,
       int M){
  int lane = threadIdx.x & 31;
  int wv   = threadIdx.x >> 5;
  int mt   = blockIdx.x*8 + wv;
  int ntm  = (M + 15) >> 4;
  if (mt >= ntm) return;                   // wave-uniform: EXEC stays all-ones for WMMA
  int hi  = lane >> 4;                     // half-wave select
  int l16 = lane & 15;
  int mrow = mt*16 + l16; if (mrow >= M) mrow = M-1;
  int ncol = (blockIdx.y << 4) + l16;
  const __bf16* arow = X  + (long)mrow*HID;
  const __bf16* brow = Wt + (long)ncol*HID;
  v8f acc = {};
  #pragma unroll
  for (int kb = 0; kb < 8; ++kb){
    int k0 = kb*32;
    union { v16bf v; v8bf h[2]; } ua, ub;
    // A 16x32: lanes 0-15 hold K = k0+[0..7], k0+16+[0..7]; lanes 16-31 shifted by 8
    ua.h[0] = *(const v8bf*)(arow + k0 + hi*8);
    ua.h[1] = *(const v8bf*)(arow + k0 + 16 + hi*8);
    // B 32x16: lanes 0-15 hold K = k0+[0..7], k0+16+[0..7]; lanes 16-31 shifted by 8
    ub.h[0] = *(const v8bf*)(brow + k0 + hi*8);
    ub.h[1] = *(const v8bf*)(brow + k0 + 16 + hi*8);
    acc = __builtin_amdgcn_wmma_f32_16x16x32_bf16(false, ua.v, false, ub.v,
                                                  (short)0, acc, false, false);
  }
  // D layout: VGPR r -> row (r + 8*hi), col l16
  int mbase = mt*16 + hi*8;
  if (Hout){
    #pragma unroll
    for (int r = 0; r < 8; ++r){
      int m = mbase + r;
      if (m < M) Hout[(long)m*HID + ncol] = acc[r];
    }
  }
  if (al_s){
    float av = avec_s[ncol];               // a_s flat (H*C) == channel index
    int hd = ncol >> 5;
    #pragma unroll
    for (int r = 0; r < 8; ++r){
      int m = mbase + r;
      if (m < M) atomicAdd(&al_s[m*NH + hd], acc[r]*av);
    }
  }
  if (al_d){
    float av = avec_d[ncol];
    int hd = ncol >> 5;
    #pragma unroll
    for (int r = 0; r < 8; ++r){
      int m = mbase + r;
      if (m < M) atomicAdd(&al_d[m*NH + hd], acc[r]*av);
    }
  }
}

// ---------------- edge passes (segment softmax + aggregation) ----------------

__global__ void k_edge_max(const int* __restrict__ ei, int E,
                           const float* __restrict__ als, const float* __restrict__ ald,
                           int* __restrict__ amax){
  int t = blockIdx.x*blockDim.x + threadIdx.x;
  if (t >= E*NH) return;
  int e = t >> 3, h = t & 7;
  int s = ei[e], d = ei[E + e];
  float a = als[s*NH + h] + ald[d*NH + h];
  a = a > 0.f ? a : SLOPE*a;
  atomicMax(&amax[d*NH + h], f2ord(a));
}

__global__ void k_edge_sum(const int* __restrict__ ei, int E,
                           const float* __restrict__ als, const float* __restrict__ ald,
                           const int* __restrict__ amax, float* __restrict__ den){
  int t = blockIdx.x*blockDim.x + threadIdx.x;
  if (t >= E*NH) return;
  int e = t >> 3, h = t & 7;
  int s = ei[e], d = ei[E + e];
  float a = als[s*NH + h] + ald[d*NH + h];
  a = a > 0.f ? a : SLOPE*a;
  float ex = expf(a - ord2f(amax[d*NH + h]));
  atomicAdd(&den[d*NH + h], ex);
}

// one wave per edge; lane l owns channels l+32*i (head i)
__global__ void __launch_bounds__(256)
k_edge_aggr(const int* __restrict__ ei, int E,
            const float* __restrict__ als, const float* __restrict__ ald,
            const int* __restrict__ amax, const float* __restrict__ den,
            const float* __restrict__ h, float* __restrict__ acc){
  int lane = threadIdx.x & 31;
  int e = blockIdx.x*(blockDim.x >> 5) + (threadIdx.x >> 5);
  if (e >= E) return;
  int s = ei[e], d = ei[E + e];
  int hh = lane & 7;                        // lanes l and l+8.. duplicate: 4x redundancy
  float a = als[s*NH + hh] + ald[d*NH + hh];
  a = a > 0.f ? a : SLOPE*a;
  float w = expf(a - ord2f(amax[d*NH + hh])) / (den[d*NH + hh] + 1e-16f);
  const float* hs = h + (long)s*HID;
  float* ac = acc + (long)d*HID;
  #pragma unroll
  for (int i = 0; i < 8; ++i){
    float wi = __shfl(w, i, 32);            // head i weight
    int ch = i*32 + lane;
    atomicAdd(&ac[ch], hs[ch]*wi);
  }
}

// ---------------- LayerNorm: one wave per node ----------------
__global__ void __launch_bounds__(256)
k_ln(const float* __restrict__ acc, const float* __restrict__ g,
     const float* __restrict__ b, float* __restrict__ out, int n){
  int lane = threadIdx.x & 31;
  int node = blockIdx.x*(blockDim.x >> 5) + (threadIdx.x >> 5);
  if (node >= n) return;
  const float* x = acc + (long)node*HID;
  float s = 0.f, s2 = 0.f;
  #pragma unroll
  for (int i = 0; i < 8; ++i){ float v = x[lane + 32*i]; s += v; s2 += v*v; }
  #pragma unroll
  for (int off = 16; off; off >>= 1){ s += __shfl_xor(s, off, 32); s2 += __shfl_xor(s2, off, 32); }
  float mu  = s  * (1.f/HID);
  float var = s2 * (1.f/HID) - mu*mu;
  float inv = rsqrtf(var + 1e-5f);
  #pragma unroll
  for (int i = 0; i < 8; ++i){
    int c = lane + 32*i;
    out[(long)node*HID + c] = (x[c] - mu)*inv*g[c] + b[c];
  }
}

// ---------------- launch ----------------
extern "C" void kernel_launch(void* const* d_in, const int* in_sizes, int n_in,
                              void* d_out, int out_size, void* d_ws, size_t ws_size,
                              hipStream_t stream){
  const float* x_gene = (const float*)d_in[0];
  const float* x_ph   = (const float*)d_in[1];
  const int*   ei_gp  = (const int*)d_in[2];
  const int*   ei_pg  = (const int*)d_in[3];
  const int*   ei_gg  = (const int*)d_in[4];
  const float* W_gp  = (const float*)d_in[5];
  const float* as_gp = (const float*)d_in[6];
  const float* ad_gp = (const float*)d_in[7];
  const float* b_gp  = (const float*)d_in[8];
  const float* W_pg  = (const float*)d_in[9];
  const float* as_pg = (const float*)d_in[10];
  const float* ad_pg = (const float*)d_in[11];
  const float* b_pg  = (const float*)d_in[12];
  const float* W_gg  = (const float*)d_in[13];
  const float* as_gg = (const float*)d_in[14];
  const float* ad_gg = (const float*)d_in[15];
  const float* b_gg  = (const float*)d_in[16];
  const float* lg_g  = (const float*)d_in[17];
  const float* lb_g  = (const float*)d_in[18];
  const float* lg_p  = (const float*)d_in[19];
  const float* lb_p  = (const float*)d_in[20];

  const int N    = in_sizes[0] / HID;
  const int E_gp = in_sizes[2] / 2;
  const int E_pg = in_sizes[3] / 2;
  const int E_gg = in_sizes[4] / 2;
  (void)n_in; (void)out_size; (void)ws_size;

  char* wp = (char*)d_ws;
  auto take = [&](size_t bytes)->void*{
    void* p = (void*)wp; wp += (bytes + 255) & ~(size_t)255; return p;
  };
  __bf16* Xg    = (__bf16*)take((size_t)N*HID*2);
  __bf16* Xp    = (__bf16*)take((size_t)N*HID*2);
  __bf16* Wt_gp = (__bf16*)take((size_t)HID*HID*2);
  __bf16* Wt_pg = (__bf16*)take((size_t)HID*HID*2);
  __bf16* Wt_gg = (__bf16*)take((size_t)HID*HID*2);
  float* h_gp = (float*)take((size_t)N*HID*4);
  float* h_pg = (float*)take((size_t)N*HID*4);
  float* h_gg = (float*)take((size_t)N*HID*4);
  // zero-init block: 6 logit arrays + 3 denominators, contiguous
  float* al_s_gp = (float*)take((size_t)N*NH*4);
  float* al_d_gp = (float*)take((size_t)N*NH*4);
  float* al_s_pg = (float*)take((size_t)N*NH*4);
  float* al_d_pg = (float*)take((size_t)N*NH*4);
  float* al_s_gg = (float*)take((size_t)N*NH*4);
  float* al_d_gg = (float*)take((size_t)N*NH*4);
  float* den_gp  = (float*)take((size_t)N*NH*4);
  float* den_pg  = (float*)take((size_t)N*NH*4);
  float* den_gg  = (float*)take((size_t)N*NH*4);
  // amax block (ordered-int encoding), contiguous
  int* amax_gp = (int*)take((size_t)N*NH*4);
  int* amax_pg = (int*)take((size_t)N*NH*4);
  int* amax_gg = (int*)take((size_t)N*NH*4);
  float* acc_g = (float*)take((size_t)N*HID*4);
  float* acc_p = (float*)take((size_t)N*HID*4);

  const int total = N*HID;
  // init: zeros for logits/denominators, INT_MIN for maxes
  {
    int nz = 9*N*NH;
    k_fill_u32<<<(nz+255)/256, 256, 0, stream>>>((unsigned*)al_s_gp, 0u, nz);
    int nm = 3*N*NH;
    k_fill_u32<<<(nm+255)/256, 256, 0, stream>>>((unsigned*)amax_gp, 0x80000000u, nm);
  }
  // fp32 -> bf16 copies of node features; W transposed+converted
  k_f2bf<<<(total+255)/256, 256, 0, stream>>>(x_gene, Xg, total);
  k_f2bf<<<(total+255)/256, 256, 0, stream>>>(x_ph,   Xp, total);
  k_wtr<<<HID*HID/256, 256, 0, stream>>>(W_gp, Wt_gp);
  k_wtr<<<HID*HID/256, 256, 0, stream>>>(W_pg, Wt_pg);
  k_wtr<<<HID*HID/256, 256, 0, stream>>>(W_gg, Wt_gg);

  // WMMA GEMMs: 5 projections (h_dst-only ones fuse into logits, never stored)
  int ntm = (N + 15) >> 4;
  dim3 ggrid((ntm + 7)/8, HID/16);
  k_gemm<<<ggrid, 256, 0, stream>>>(Xg, Wt_gp, h_gp,   as_gp, al_s_gp, nullptr, nullptr, N);
  k_gemm<<<ggrid, 256, 0, stream>>>(Xp, Wt_gp, nullptr, nullptr, nullptr, ad_gp, al_d_gp, N);
  k_gemm<<<ggrid, 256, 0, stream>>>(Xp, Wt_pg, h_pg,   as_pg, al_s_pg, nullptr, nullptr, N);
  k_gemm<<<ggrid, 256, 0, stream>>>(Xg, Wt_pg, nullptr, nullptr, nullptr, ad_pg, al_d_pg, N);
  k_gemm<<<ggrid, 256, 0, stream>>>(Xg, Wt_gg, h_gg,   as_gg, al_s_gg, ad_gg, al_d_gg, N);

  // accumulators = residual + biases (HeteroConv aggr='sum')
  k_init_acc<<<(total+255)/256, 256, 0, stream>>>(x_gene, b_pg, b_gg, acc_g, total);
  k_init_acc<<<(total+255)/256, 256, 0, stream>>>(x_ph,   b_gp, nullptr, acc_p, total);

  // segment softmax: max pass
  k_edge_max<<<(E_gp*NH+255)/256, 256, 0, stream>>>(ei_gp, E_gp, al_s_gp, al_d_gp, amax_gp);
  k_edge_max<<<(E_pg*NH+255)/256, 256, 0, stream>>>(ei_pg, E_pg, al_s_pg, al_d_pg, amax_pg);
  k_edge_max<<<(E_gg*NH+255)/256, 256, 0, stream>>>(ei_gg, E_gg, al_s_gg, al_d_gg, amax_gg);
  // denominator pass
  k_edge_sum<<<(E_gp*NH+255)/256, 256, 0, stream>>>(ei_gp, E_gp, al_s_gp, al_d_gp, amax_gp, den_gp);
  k_edge_sum<<<(E_pg*NH+255)/256, 256, 0, stream>>>(ei_pg, E_pg, al_s_pg, al_d_pg, amax_pg, den_pg);
  k_edge_sum<<<(E_gg*NH+255)/256, 256, 0, stream>>>(ei_gg, E_gg, al_s_gg, al_d_gg, amax_gg, den_gg);
  // weighted message aggregation (memory-bound pass), one wave per edge
  k_edge_aggr<<<(E_gp+7)/8, 256, 0, stream>>>(ei_gp, E_gp, al_s_gp, al_d_gp, amax_gp, den_gp, h_gp, acc_p);
  k_edge_aggr<<<(E_pg+7)/8, 256, 0, stream>>>(ei_pg, E_pg, al_s_pg, al_d_pg, amax_pg, den_pg, h_pg, acc_g);
  k_edge_aggr<<<(E_gg+7)/8, 256, 0, stream>>>(ei_gg, E_gg, al_s_gg, al_d_gg, amax_gg, den_gg, h_gg, acc_g);

  // residual already in acc; LayerNorm to outputs (gene first, then pheno)
  float* out = (float*)d_out;
  k_ln<<<(N+7)/8, 256, 0, stream>>>(acc_g, lg_g, lb_g, out, N);
  k_ln<<<(N+7)/8, 256, 0, stream>>>(acc_p, lg_p, lb_p, out + (size_t)N*HID, N);
}
